// MSDeformAttnTransformerEncoderOnly_6906307412625
// MI455X (gfx1250) — matrix operand order repaired and checked by reading, not compiled, not run on hardware
//
#include <hip/hip_runtime.h>

// ---------------------------------------------------------------------------
// MS-Deformable-Attention Transformer Encoder for MI455X (gfx1250, wave32)
// GEMMs via V_WMMA_F32_16X16X32_BF16 (f32 accumulate). Block tile 256x64,
// 8 waves x (32x64) each: B fragments reused across two M-subtiles ->
// 1.5 ds_load_b128 per WMMA. Staging loads kept live in distinct registers
// via a scalar-operand asm fence so all 5 global_load_b128 pipeline together.
// ---------------------------------------------------------------------------

#define D_MODEL   256
#define N_HEADS   8
#define HEAD_DIM  32
#define N_LEVELS  4
#define N_POINTS  4
#define N_LAYERS  6
#define D_FF      1024
#define BATCH     2
#define NTOK      21760                 // 128^2 + 64^2 + 32^2 + 16^2
#define ROWS      (BATCH * NTOK)        // 43520 = 170 * 256

typedef __attribute__((ext_vector_type(16))) __bf16 v16bf;
typedef __attribute__((ext_vector_type(8)))  float  v8f;

union FragBF { uint4 u[2]; v16bf v; };

__device__ __forceinline__ unsigned short f2bf(float f) {
  unsigned int u = __float_as_uint(f);
  u += 0x7FFFu + ((u >> 16) & 1u);      // round-to-nearest-even
  return (unsigned short)(u >> 16);
}

__device__ __forceinline__ void lvl_decode(int n, int& lvl, int& px, int& wn) {
  if (n < 16384)      { lvl = 0; px = n;         wn = 128; }
  else if (n < 20480) { lvl = 1; px = n - 16384; wn = 64;  }
  else if (n < 21504) { lvl = 2; px = n - 20480; wn = 32;  }
  else                { lvl = 3; px = n - 21504; wn = 16;  }
}

// ---------------------------------------------------------------------------
// Weight prep: fp32 [L][K][F] -> bf16 transposed [L][F][K]
// ---------------------------------------------------------------------------
__global__ void tcast_kernel(const float* __restrict__ src,
                             unsigned short* __restrict__ dst,
                             int K, int F, int L) {
  size_t total = (size_t)L * K * F;
  size_t kf = (size_t)K * F;
  for (size_t i = (size_t)blockIdx.x * blockDim.x + threadIdx.x; i < total;
       i += (size_t)gridDim.x * blockDim.x) {
    size_t l = i / kf;
    size_t r = i - l * kf;
    int k = (int)(r / F);
    int f = (int)(r - (size_t)k * F);
    dst[l * kf + (size_t)f * K + k] = f2bf(src[i]);
  }
}

// ---------------------------------------------------------------------------
// Pack: src_lvl [B][256][h][w] -> q [B*N][256] ; pos + level_embed -> qpos
// ---------------------------------------------------------------------------
__global__ __launch_bounds__(256)
void pack_q_kernel(const float* __restrict__ s0, const float* __restrict__ s1,
                   const float* __restrict__ s2, const float* __restrict__ s3,
                   float* __restrict__ q) {
  const int row = blockIdx.x;           // b*NTOK + n
  const int c   = threadIdx.x;
  const int b   = row / NTOK;
  const int n   = row - b * NTOK;
  int lvl, px, wn;
  lvl_decode(n, lvl, px, wn);
  const int hw = wn * wn;
  const float* s = (lvl == 0) ? s0 : (lvl == 1) ? s1 : (lvl == 2) ? s2 : s3;
  q[(size_t)row * D_MODEL + c] = s[(size_t)(b * D_MODEL + c) * hw + px];
}

__global__ __launch_bounds__(256)
void pack_qpos_kernel(const float* __restrict__ p0, const float* __restrict__ p1,
                      const float* __restrict__ p2, const float* __restrict__ p3,
                      const float* __restrict__ level_embed,
                      float* __restrict__ qpos) {
  const int row = blockIdx.x;
  const int c   = threadIdx.x;
  const int b   = row / NTOK;
  const int n   = row - b * NTOK;
  int lvl, px, wn;
  lvl_decode(n, lvl, px, wn);
  const int hw = wn * wn;
  const float* p = (lvl == 0) ? p0 : (lvl == 1) ? p1 : (lvl == 2) ? p2 : p3;
  qpos[(size_t)row * D_MODEL + c] =
      p[(size_t)(b * D_MODEL + c) * hw + px] + level_embed[lvl * D_MODEL + c];
}

// ---------------------------------------------------------------------------
// fp32 -> bf16 casts
// ---------------------------------------------------------------------------
__global__ void castbf_kernel(const float* __restrict__ a,
                              unsigned short* __restrict__ d, size_t n) {
  for (size_t i = (size_t)blockIdx.x * blockDim.x + threadIdx.x; i < n;
       i += (size_t)gridDim.x * blockDim.x)
    d[i] = f2bf(a[i]);
}

__global__ void addcastbf_kernel(const float* __restrict__ a,
                                 const float* __restrict__ b,
                                 unsigned short* __restrict__ d, size_t n) {
  for (size_t i = (size_t)blockIdx.x * blockDim.x + threadIdx.x; i < n;
       i += (size_t)gridDim.x * blockDim.x)
    d[i] = f2bf(a[i] + b[i]);
}

// ---------------------------------------------------------------------------
// bf16 WMMA GEMM: C[rows x F] = A[rows x K] (bf16) * Wt[F x K]^T (bf16) + bias
// Block: 256 thr = 8 waves; tile 256x64; each wave -> 32 rows x 64 cols
// (two 16-row M-subtiles sharing B fragments). EPI 0: fp32; EPI 1: relu->bf16.
// ---------------------------------------------------------------------------
#define BM 256
#define BN 64
#define BK 32
#define LDS_STRIDE 40   // ushort elems / row (80B: 16B-aligned, conflict-free)

template <int EPI>
__global__ __launch_bounds__(256)
void gemm_bf16_kernel(const unsigned short* __restrict__ A,
                      const unsigned short* __restrict__ Wt,
                      const float* __restrict__ bias,
                      void* __restrict__ Cout,
                      int rows, int K, int F) {
  __shared__ __align__(16) unsigned short lA[BM * LDS_STRIDE];  // 20480 B
  __shared__ __align__(16) unsigned short lB[BN * LDS_STRIDE];  //  5120 B

  const int tid  = threadIdx.x;
  const int wave = tid >> 5;
  const int lane = tid & 31;
  const int r0   = blockIdx.x * BM;
  const int c0   = blockIdx.y * BN;

  // Per-thread staging coordinates (fixed across K-steps)
  int arow[4], acb[4];
#pragma unroll
  for (int i = 0; i < 4; ++i) {
    const int chunk = tid + i * 256;              // 0..1023
    arow[i] = chunk >> 2;
    acb[i]  = (chunk & 3) * 8;                    // 0,8,16,24 elems
  }
  const int brow = tid >> 2;
  const int bcb  = (tid & 3) * 8;

  v8f zero = {0.f, 0.f, 0.f, 0.f, 0.f, 0.f, 0.f, 0.f};
  v8f acc[2][4];
#pragma unroll
  for (int mt = 0; mt < 2; ++mt)
#pragma unroll
    for (int nt = 0; nt < 4; ++nt) acc[mt][nt] = zero;

  for (int k0 = 0; k0 < K; k0 += BK) {
    // ---- Stage: issue all 5 global b128 loads, keep in distinct registers ----
    uint4 ra0 = *(const uint4*)(A + (size_t)(r0 + arow[0]) * K + k0 + acb[0]);
    uint4 ra1 = *(const uint4*)(A + (size_t)(r0 + arow[1]) * K + k0 + acb[1]);
    uint4 ra2 = *(const uint4*)(A + (size_t)(r0 + arow[2]) * K + k0 + acb[2]);
    uint4 ra3 = *(const uint4*)(A + (size_t)(r0 + arow[3]) * K + k0 + acb[3]);
    uint4 rb  = *(const uint4*)(Wt + (size_t)(c0 + brow) * K + k0 + bcb);
    // Scheduling fence (input-only 32-bit operands, supported by the backend):
    // all 20 dwords must be live at once -> regalloc cannot coalesce the five
    // loads into one quad, so they pipeline instead of draining one-by-one.
    asm volatile("" ::
        "v"(ra0.x), "v"(ra0.y), "v"(ra0.z), "v"(ra0.w),
        "v"(ra1.x), "v"(ra1.y), "v"(ra1.z), "v"(ra1.w),
        "v"(ra2.x), "v"(ra2.y), "v"(ra2.z), "v"(ra2.w),
        "v"(ra3.x), "v"(ra3.y), "v"(ra3.z), "v"(ra3.w),
        "v"(rb.x),  "v"(rb.y),  "v"(rb.z),  "v"(rb.w));
    // ---- commit to LDS ----
    *(uint4*)(&lA[arow[0] * LDS_STRIDE + acb[0]]) = ra0;
    *(uint4*)(&lA[arow[1] * LDS_STRIDE + acb[1]]) = ra1;
    *(uint4*)(&lA[arow[2] * LDS_STRIDE + acb[2]]) = ra2;
    *(uint4*)(&lA[arow[3] * LDS_STRIDE + acb[3]]) = ra3;
    *(uint4*)(&lB[brow * LDS_STRIDE + bcb]) = rb;

    // Prefetch next K-slice (global_prefetch_b8)
    if (k0 + BK < K) {
      __builtin_prefetch(A + (size_t)(r0 + tid) * K + k0 + BK, 0, 1);
      __builtin_prefetch(Wt + (size_t)(c0 + (tid & 63)) * K + k0 + BK, 0, 1);
    }
    __syncthreads();

    // A fragments for two M-subtiles:
    // lane<16 holds K{0..7,16..23}, lane>=16 K{8..15,24..31}
    const int ka = (lane < 16) ? 0 : 8;
    FragBF fa[2];
#pragma unroll
    for (int mt = 0; mt < 2; ++mt) {
      const int mrow = wave * 32 + mt * 16 + (lane & 15);
      fa[mt].u[0] = *(const uint4*)(&lA[mrow * LDS_STRIDE + ka]);
      fa[mt].u[1] = *(const uint4*)(&lA[mrow * LDS_STRIDE + 16 + ka]);
    }

#pragma unroll
    for (int nt = 0; nt < 4; ++nt) {
      // B fragment: lane<16 holds K{0..7,16..23}, lane>=16 K{8..15,24..31} of column n=lane%16
      FragBF fb;
      const int nrow = nt * 16 + (lane & 15);
      const int kb   = (lane < 16) ? 0 : 8;
      fb.u[0] = *(const uint4*)(&lB[nrow * LDS_STRIDE + kb]);
      fb.u[1] = *(const uint4*)(&lB[nrow * LDS_STRIDE + kb + 16]);
#pragma unroll
      for (int mt = 0; mt < 2; ++mt)
        acc[mt][nt] = __builtin_amdgcn_wmma_f32_16x16x32_bf16(
            false, fa[mt].v, false, fb.v, (short)0, acc[mt][nt], false, false);
    }
    __syncthreads();
  }

  // Epilogue: C/D layout — VGPR r: M = r + (lane<16 ? 0 : 8), N = lane%16
  const int nIn = lane & 15;
#pragma unroll
  for (int mt = 0; mt < 2; ++mt) {
    const int mBase = r0 + wave * 32 + mt * 16 + ((lane >> 4) << 3);
#pragma unroll
    for (int nt = 0; nt < 4; ++nt) {
      const int col = c0 + nt * 16 + nIn;
      const float bcol = bias[col];
#pragma unroll
      for (int r = 0; r < 8; ++r) {
        float v = acc[mt][nt][r] + bcol;
        if (EPI == 1) {
          v = v > 0.f ? v : 0.f;
          ((unsigned short*)Cout)[(size_t)(mBase + r) * F + col] = f2bf(v);
        } else {
          ((float*)Cout)[(size_t)(mBase + r) * F + col] = v;
        }
      }
    }
  }
}

// ---------------------------------------------------------------------------
// Deformable attention sampling: one wave per (b, token, head); lane = channel
// ---------------------------------------------------------------------------
__global__ __launch_bounds__(256)
void msdeform_attn_kernel(const float* __restrict__ value,  // [B*N][256]
                          const float* __restrict__ off,    // [B*N][256]
                          const float* __restrict__ aw,     // [B*N][128]
                          float* __restrict__ out) {        // [B*N][256]
  const int unit = blockIdx.x * 8 + (threadIdx.x >> 5);
  const int lane = threadIdx.x & 31;
  const int h    = unit & 7;
  const int row  = unit >> 3;            // b*NTOK + n
  const int n    = row % NTOK;
  const int rowBase = row - n;           // b*NTOK

  int lvl, px, wn;
  lvl_decode(n, lvl, px, wn);
  const float refx = ((float)(px & (wn - 1)) + 0.5f) / (float)wn;
  const float refy = ((float)(px / wn) + 0.5f) / (float)wn;

  const float* awp  = aw  + (size_t)row * 128 + h * 16;
  const float* offp = off + (size_t)row * 256 + h * 32;

  float mx = -1e30f;
#pragma unroll
  for (int j = 0; j < 16; ++j) mx = fmaxf(mx, awp[j]);
  float sum = 0.f;
#pragma unroll
  for (int j = 0; j < 16; ++j) sum += __expf(awp[j] - mx);
  const float inv = 1.f / sum;

  const int startT[4] = {0, 16384, 20480, 21504};
  float acc = 0.f;
#pragma unroll
  for (int l = 0; l < 4; ++l) {
    const int wl = 128 >> l;
    const float fwl = (float)wl;
    const size_t vbase =
        (size_t)(rowBase + startT[l]) * D_MODEL + h * HEAD_DIM + lane;
#pragma unroll
    for (int p = 0; p < 4; ++p) {
      const float ox = offp[(l * 4 + p) * 2 + 0];
      const float oy = offp[(l * 4 + p) * 2 + 1];
      const float fx = (refx + ox / fwl) * fwl - 0.5f;
      const float fy = (refy + oy / fwl) * fwl - 0.5f;
      const float x0f = floorf(fx), y0f = floorf(fy);
      const float dx = fx - x0f, dy = fy - y0f;
      const int x0 = (int)x0f, y0 = (int)y0f;
      float s = 0.f;
#pragma unroll
      for (int cy = 0; cy < 2; ++cy)
#pragma unroll
        for (int cx = 0; cx < 2; ++cx) {
          const int xi = x0 + cx, yi = y0 + cy;
          if (xi >= 0 && xi < wl && yi >= 0 && yi < wl) {
            const float wgt = (cx ? dx : 1.f - dx) * (cy ? dy : 1.f - dy);
            s += wgt * value[vbase + (size_t)(yi * wl + xi) * D_MODEL];
          }
        }
      acc += __expf(awp[l * 4 + p] - mx) * inv * s;
    }
  }
  out[(size_t)row * D_MODEL + h * HEAD_DIM + lane] = acc;
}

// ---------------------------------------------------------------------------
// q = layernorm(q + delta) ; one wave per row, 8 channels per lane (wave32)
// ---------------------------------------------------------------------------
__global__ __launch_bounds__(256)
void residual_ln_kernel(float* __restrict__ q, const float* __restrict__ delta,
                        const float* __restrict__ g, const float* __restrict__ b) {
  const int row  = blockIdx.x * 8 + (threadIdx.x >> 5);
  const int lane = threadIdx.x & 31;
  float x[8];
  float s = 0.f;
#pragma unroll
  for (int i = 0; i < 8; ++i) {
    const int c = lane + i * 32;
    x[i] = q[(size_t)row * D_MODEL + c] + delta[(size_t)row * D_MODEL + c];
    s += x[i];
  }
#pragma unroll
  for (int o = 16; o > 0; o >>= 1) s += __shfl_xor(s, o, 32);
  const float mu = s * (1.f / D_MODEL);
  float v = 0.f;
#pragma unroll
  for (int i = 0; i < 8; ++i) { const float d = x[i] - mu; v += d * d; }
#pragma unroll
  for (int o = 16; o > 0; o >>= 1) v += __shfl_xor(v, o, 32);
  const float rinv = rsqrtf(v * (1.f / D_MODEL) + 1e-5f);
#pragma unroll
  for (int i = 0; i < 8; ++i) {
    const int c = lane + i * 32;
    q[(size_t)row * D_MODEL + c] = (x[i] - mu) * rinv * g[c] + b[c];
  }
}

__global__ void copyf_kernel(const float* __restrict__ s, float* __restrict__ d,
                             size_t n) {
  for (size_t i = (size_t)blockIdx.x * blockDim.x + threadIdx.x; i < n;
       i += (size_t)gridDim.x * blockDim.x)
    d[i] = s[i];
}

// ---------------------------------------------------------------------------
// Host orchestration
// ---------------------------------------------------------------------------
extern "C" void kernel_launch(void* const* d_in, const int* in_sizes, int n_in,
                              void* d_out, int out_size, void* d_ws, size_t ws_size,
                              hipStream_t stream) {
  (void)in_sizes; (void)n_in; (void)out_size; (void)ws_size;
  // setup_inputs() dict order: src0,pos0,src1,pos1,src2,pos2,src3,pos3, then params
  const float* src0 = (const float*)d_in[0];
  const float* pos0 = (const float*)d_in[1];
  const float* src1 = (const float*)d_in[2];
  const float* pos1 = (const float*)d_in[3];
  const float* src2 = (const float*)d_in[4];
  const float* pos2 = (const float*)d_in[5];
  const float* src3 = (const float*)d_in[6];
  const float* pos3 = (const float*)d_in[7];
  const float* level_embed = (const float*)d_in[8];
  const float* Wv  = (const float*)d_in[9];
  const float* bv  = (const float*)d_in[10];
  const float* Ws  = (const float*)d_in[11];
  const float* bso = (const float*)d_in[12];
  const float* Wa  = (const float*)d_in[13];
  const float* ba  = (const float*)d_in[14];
  const float* Wo  = (const float*)d_in[15];
  const float* bo  = (const float*)d_in[16];
  const float* g1  = (const float*)d_in[17];
  const float* be1 = (const float*)d_in[18];
  const float* W1  = (const float*)d_in[19];
  const float* b1  = (const float*)d_in[20];
  const float* W2  = (const float*)d_in[21];
  const float* b2  = (const float*)d_in[22];
  const float* g2  = (const float*)d_in[23];
  const float* be2 = (const float*)d_in[24];

  char* p = (char*)d_ws;
  auto alloc = [&](size_t bytes) -> void* {
    void* r = (void*)p;
    p += (bytes + 255) & ~(size_t)255;
    return r;
  };
  const size_t NE  = (size_t)ROWS * D_MODEL;     // 11,141,120
  float*          qF     = (float*)alloc(NE * 4);
  float*          qposF  = (float*)alloc(NE * 4);
  float*          valueF = (float*)alloc(NE * 4);
  float*          offF   = (float*)alloc(NE * 4);
  float*          awF    = (float*)alloc((size_t)ROWS * 128 * 4);
  float*          attnF  = (float*)alloc(NE * 4);
  float*          tmpF   = (float*)alloc(NE * 4);
  unsigned short* qBf    = (unsigned short*)alloc(NE * 2);
  unsigned short* qpBf   = (unsigned short*)alloc(NE * 2);
  unsigned short* attnBf = (unsigned short*)alloc(NE * 2);
  unsigned short* ffnBf  = (unsigned short*)alloc((size_t)ROWS * D_FF * 2);
  unsigned short* wWv = (unsigned short*)alloc((size_t)N_LAYERS * 256 * 256 * 2);
  unsigned short* wWs = (unsigned short*)alloc((size_t)N_LAYERS * 256 * 256 * 2);
  unsigned short* wWa = (unsigned short*)alloc((size_t)N_LAYERS * 256 * 128 * 2);
  unsigned short* wWo = (unsigned short*)alloc((size_t)N_LAYERS * 256 * 256 * 2);
  unsigned short* wW1 = (unsigned short*)alloc((size_t)N_LAYERS * 256 * 1024 * 2);
  unsigned short* wW2 = (unsigned short*)alloc((size_t)N_LAYERS * 1024 * 256 * 2);

  // Weight prep (cast + transpose to [F][K]) — deterministic each call
  tcast_kernel<<<2048, 256, 0, stream>>>(Wv, wWv, 256, 256, N_LAYERS);
  tcast_kernel<<<2048, 256, 0, stream>>>(Ws, wWs, 256, 256, N_LAYERS);
  tcast_kernel<<<1024, 256, 0, stream>>>(Wa, wWa, 256, 128, N_LAYERS);
  tcast_kernel<<<2048, 256, 0, stream>>>(Wo, wWo, 256, 256, N_LAYERS);
  tcast_kernel<<<4096, 256, 0, stream>>>(W1, wW1, 256, 1024, N_LAYERS);
  tcast_kernel<<<4096, 256, 0, stream>>>(W2, wW2, 1024, 256, N_LAYERS);

  pack_q_kernel<<<ROWS, 256, 0, stream>>>(src0, src1, src2, src3, qF);
  pack_qpos_kernel<<<ROWS, 256, 0, stream>>>(pos0, pos1, pos2, pos3, level_embed,
                                             qposF);

  const dim3 g256(ROWS / BM, D_MODEL / BN);   // (170, 4)
  const dim3 g128(ROWS / BM, 128 / BN);       // (170, 2)
  const dim3 g1024(ROWS / BM, D_FF / BN);     // (170, 16)

  for (int l = 0; l < N_LAYERS; ++l) {
    castbf_kernel<<<4096, 256, 0, stream>>>(qF, qBf, NE);
    addcastbf_kernel<<<4096, 256, 0, stream>>>(qF, qposF, qpBf, NE);

    gemm_bf16_kernel<0><<<g256, 256, 0, stream>>>(
        qBf, wWv + (size_t)l * 256 * 256, bv + l * 256, valueF, ROWS, 256, 256);
    gemm_bf16_kernel<0><<<g256, 256, 0, stream>>>(
        qpBf, wWs + (size_t)l * 256 * 256, bso + l * 256, offF, ROWS, 256, 256);
    gemm_bf16_kernel<0><<<g128, 256, 0, stream>>>(
        qpBf, wWa + (size_t)l * 256 * 128, ba + l * 128, awF, ROWS, 256, 128);

    msdeform_attn_kernel<<<ROWS, 256, 0, stream>>>(valueF, offF, awF, attnF);

    castbf_kernel<<<4096, 256, 0, stream>>>(attnF, attnBf, NE);
    gemm_bf16_kernel<0><<<g256, 256, 0, stream>>>(
        attnBf, wWo + (size_t)l * 256 * 256, bo + l * 256, tmpF, ROWS, 256, 256);

    residual_ln_kernel<<<ROWS / 8, 256, 0, stream>>>(qF, tmpF, g1 + l * 256,
                                                     be1 + l * 256);

    castbf_kernel<<<4096, 256, 0, stream>>>(qF, qBf, NE);
    gemm_bf16_kernel<1><<<g1024, 256, 0, stream>>>(
        qBf, wW1 + (size_t)l * 256 * 1024, b1 + l * 1024, ffnBf, ROWS, 256, 1024);
    gemm_bf16_kernel<0><<<g256, 256, 0, stream>>>(
        ffnBf, wW2 + (size_t)l * 1024 * 256, b2 + l * 256, tmpF, ROWS, 1024, 256);

    residual_ln_kernel<<<ROWS / 8, 256, 0, stream>>>(qF, tmpF, g2 + l * 256,
                                                     be2 + l * 256);
  }

  copyf_kernel<<<4096, 256, 0, stream>>>(qF, (float*)d_out, NE);
}
